// HierarchicalGATNet_10771777979029
// MI455X (gfx1250) — matrix operand order repaired and checked by reading, not compiled, not run on hardware
//
#include <hip/hip_runtime.h>
#include <hip/hip_bf16.h>
#include <math.h>

// ---------------------------------------------------------------------------
// HierarchicalGATNet on gfx1250 (MI455X).
//  - Dense feature transforms: f32 -> f16 pack into WMMA fragment layout,
//    then V_WMMA_F32_16X16X32_F16 with a branch-free, fully-unrolled K loop
//    and a guard-free store epilogue (outputs padded to 16x16 tiles).
//  - Edge softmax / aggregation: thread-per-edge kernels + float atomics
//    (N*H softmax state fits easily in the 192 MB L2).
// ---------------------------------------------------------------------------

typedef __attribute__((ext_vector_type(16))) _Float16 v16h;
typedef __attribute__((ext_vector_type(8)))  float    v8f;

#define EPS 1e-16f

__device__ __forceinline__ float leaky02(float v) {
    return v > 0.0f ? v : 0.2f * v;
}

__device__ __forceinline__ void atomicMaxF(float* addr, float val) {
    int* ia = reinterpret_cast<int*>(addr);
    int cur = __float_as_int(*addr);
    while (__int_as_float(cur) < val) {
        int prev = atomicCAS(ia, cur, __float_as_int(val));
        if (prev == cur) break;
        cur = prev;
    }
}

// ---------------------------------------------------------------------------
// BFS hierarchy
// ---------------------------------------------------------------------------
__global__ void k_bfs_init(const int* __restrict__ seed_mask,
                           int* __restrict__ dist, int* __restrict__ reached,
                           int* __restrict__ nbr, int n) {
    int i = blockIdx.x * blockDim.x + threadIdx.x;
    if (i >= n) return;
    int r = (seed_mask[i] == 0) ? 1 : 0;
    reached[i] = r;
    dist[i] = r ? 0 : 4;   // MAX_HOPS + 1
    nbr[i] = 0;
}

__global__ void k_bfs_edge(const long long* __restrict__ src,
                           const long long* __restrict__ dst,
                           const int* __restrict__ reached,
                           int* __restrict__ nbr, int ne) {
    int e = blockIdx.x * blockDim.x + threadIdx.x;
    if (e >= ne) return;
    int s = (int)src[e];
    if (reached[s]) nbr[(int)dst[e]] = 1;   // idempotent store of 1
}

__global__ void k_bfs_node(int* __restrict__ dist, int* __restrict__ reached,
                           int* __restrict__ nbr, int hop, int n) {
    int i = blockIdx.x * blockDim.x + threadIdx.x;
    if (i >= n) return;
    if (nbr[i] && !reached[i]) { dist[i] = hop; reached[i] = 1; }
    nbr[i] = 0;
}

// aug[:, :128] = x ; aug[:, 128:132] = one-hot(dist) (zero row if dist > 3)
__global__ void k_build_aug(const float* __restrict__ x,
                            const int* __restrict__ dist,
                            float* __restrict__ aug, int n) {
    long long idx = (long long)blockIdx.x * blockDim.x + threadIdx.x;
    long long total = (long long)n * 132;
    if (idx >= total) return;
    int i = (int)(idx / 132);
    int c = (int)(idx % 132);
    float v;
    if (c < 128) v = x[(long long)i * 128 + c];
    else         v = (dist[i] == (c - 128)) ? 1.0f : 0.0f;
    aug[idx] = v;
}

__global__ void k_hop(const int* __restrict__ dist, float* __restrict__ hop, int n) {
    int i = blockIdx.x * blockDim.x + threadIdx.x;
    if (i >= n) return;
    int d = dist[i];
    hop[i] = (d <= 3) ? (float)d : 0.0f;   // argmax of all-zero row -> 0
}

// ---------------------------------------------------------------------------
// Fragment packing (CDNA5 ISA 7.12.2 layouts), f32 -> f16, zero padded.
//
// A (16x32 per tile): lane = hi*16 + mloc ; half j:
//     m = tile_m*16 + mloc ;  k = kt + hi*8 + (j<8 ? j : 16 + (j-8))
// B (32x16 per tile): lane = hi*16 + nloc ; half j:
//     n = tile_n*16 + nloc ;  k = kt + hi*8 + (j<8 ? j : 16 + (j-8))
// Packed storage: Ap[((tile_m*KT + kt_i)*32 + lane)*16 + j]
// ---------------------------------------------------------------------------
__global__ void k_pack_a(const float* __restrict__ A, _Float16* __restrict__ Ap,
                         int M, int K, int KT) {
    long long idx = (long long)blockIdx.x * blockDim.x + threadIdx.x;
    long long Mt = (M + 15) / 16;
    long long total = Mt * KT * 512;
    if (idx >= total) return;
    int j    = (int)(idx & 15);
    int lane = (int)((idx >> 4) & 31);
    long long tile = idx >> 9;
    int kti = (int)(tile % KT);
    int tm  = (int)(tile / KT);
    int mloc = lane & 15, hi = lane >> 4;
    int m = tm * 16 + mloc;
    int k = kti * 32 + hi * 8 + (j < 8 ? j : 16 + (j - 8));
    float v = (m < M && k < K) ? A[(long long)m * K + k] : 0.0f;
    Ap[idx] = (_Float16)v;
}

__global__ void k_pack_b(const float* __restrict__ B, _Float16* __restrict__ Bp,
                         int K, int NC, int KT) {
    long long idx = (long long)blockIdx.x * blockDim.x + threadIdx.x;
    long long Nt = (NC + 15) / 16;
    long long total = Nt * KT * 512;
    if (idx >= total) return;
    int j    = (int)(idx & 15);
    int lane = (int)((idx >> 4) & 31);
    long long tile = idx >> 9;
    int kti = (int)(tile % KT);
    int tn  = (int)(tile / KT);
    int nloc = lane & 15, hi = lane >> 4;
    int n = tn * 16 + nloc;
    int k = kti * 32 + hi * 8 + (j < 8 ? j : 16 + (j - 8));
    float v = (k < K && n < NC) ? B[(long long)k * NC + n] : 0.0f;
    Bp[idx] = (_Float16)v;
}

// ---------------------------------------------------------------------------
// WMMA GEMM on packed fragments: C[Mt*16, ldc] = A * B (f16 mul / f32 acc).
// One wave per 16x16 output tile; K loop fully unrolled (templated KT).
// Output buffer is padded to whole tiles (ldc = NT*16, Mt*16 rows), so the
// epilogue is 8 unconditional coalesced stores - no exec-mask churn.
// ---------------------------------------------------------------------------
template <int KT>
__global__ __launch_bounds__(32)
void k_wmma_gemm(const _Float16* __restrict__ Ap, const _Float16* __restrict__ Bp,
                 float* __restrict__ C, int ldc) {
    int tm   = blockIdx.x;
    int tn   = blockIdx.y;
    int lane = threadIdx.x;

    const v16h* ap = reinterpret_cast<const v16h*>(Ap) + ((long long)tm * KT) * 32 + lane;
    const v16h* bp = reinterpret_cast<const v16h*>(Bp) + ((long long)tn * KT) * 32 + lane;

    v8f acc = {};
#pragma unroll
    for (int kt = 0; kt < KT; ++kt) {
        v16h a = ap[(long long)kt * 32];
        v16h b = bp[(long long)kt * 32];
        acc = __builtin_amdgcn_wmma_f32_16x16x32_f16(
            /*neg_a=*/false, a, /*neg_b=*/false, b,
            /*c_mod=*/(short)0, acc, /*reuse_a=*/false, /*reuse_b=*/false);
    }

    int mloc = lane & 15, hi = lane >> 4;
    float* cbase = C + (long long)(tm * 16 + hi * 8) * ldc + tn * 16 + mloc;
#pragma unroll
    for (int r = 0; r < 8; ++r) {
        cbase[(long long)r * ldc] = acc[r];
    }
}

// ---------------------------------------------------------------------------
// GAT attention pieces (generic over H heads, C channels).
// ht rows have leading dimension ldh (>= H*C; 48 for the 40-col layer).
// ---------------------------------------------------------------------------
__global__ void k_attn_logits(const float* __restrict__ ht,
                              const float* __restrict__ a_src,
                              const float* __restrict__ a_dst,
                              float* __restrict__ als, float* __restrict__ ald,
                              int n, int H, int C, int ldh) {
    int i = blockIdx.x * blockDim.x + threadIdx.x;
    if (i >= n) return;
    const float* hrow = ht + (long long)i * ldh;
    for (int h = 0; h < H; ++h) {
        float ss = 0.0f, sd = 0.0f;
        for (int c = 0; c < C; ++c) {
            float v = hrow[h * C + c];
            ss += v * a_src[h * C + c];
            sd += v * a_dst[h * C + c];
        }
        als[i * H + h] = ss;
        ald[i * H + h] = sd;
    }
}

// seed running max with the self-loop logit (self loops exist for every node)
__global__ void k_attn_init(const float* __restrict__ als,
                            const float* __restrict__ ald,
                            float* __restrict__ emax, int n, int H) {
    int i = blockIdx.x * blockDim.x + threadIdx.x;
    if (i >= n) return;
    for (int h = 0; h < H; ++h)
        emax[i * H + h] = leaky02(als[i * H + h] + ald[i * H + h]);
}

__global__ void k_attn_edge_max(const long long* __restrict__ src,
                                const long long* __restrict__ dst,
                                const float* __restrict__ als,
                                const float* __restrict__ ald,
                                float* __restrict__ emax, int ne, int H) {
    int e = blockIdx.x * blockDim.x + threadIdx.x;
    if (e >= ne) return;
    int s = (int)src[e], d = (int)dst[e];
    for (int h = 0; h < H; ++h) {
        float v = leaky02(als[s * H + h] + ald[d * H + h]);
        atomicMaxF(&emax[d * H + h], v);
    }
}

// self-loop contribution seeds denom and accumulator (no memset needed)
__global__ void k_attn_self(const float* __restrict__ ht,
                            const float* __restrict__ als,
                            const float* __restrict__ ald,
                            const float* __restrict__ emax,
                            float* __restrict__ eden, float* __restrict__ acc,
                            int n, int H, int C, int ldh) {
    int i = blockIdx.x * blockDim.x + threadIdx.x;
    if (i >= n) return;
    for (int h = 0; h < H; ++h) {
        float v = leaky02(als[i * H + h] + ald[i * H + h]);
        float w = __expf(v - emax[i * H + h]);   // <= 1 by construction
        eden[i * H + h] = w;
        const float* hrow = ht + (long long)i * ldh + h * C;
        float* arow = acc + ((long long)i * H + h) * C;
        for (int c = 0; c < C; ++c) arow[c] = w * hrow[c];
    }
}

__global__ void k_attn_edge_sum(const long long* __restrict__ src,
                                const long long* __restrict__ dst,
                                const float* __restrict__ ht,
                                const float* __restrict__ als,
                                const float* __restrict__ ald,
                                const float* __restrict__ emax,
                                float* __restrict__ eden, float* __restrict__ acc,
                                int ne, int H, int C, int ldh) {
    int e = blockIdx.x * blockDim.x + threadIdx.x;
    if (e >= ne) return;
    int s = (int)src[e], d = (int)dst[e];
    for (int h = 0; h < H; ++h) {
        float v = leaky02(als[s * H + h] + ald[d * H + h]);
        float w = __expf(v - emax[d * H + h]);
        atomicAdd(&eden[d * H + h], w);
        const float* hrow = ht + (long long)s * ldh + h * C;
        float* arow = acc + ((long long)d * H + h) * C;
        for (int c = 0; c < C; ++c) atomicAdd(&arow[c], w * hrow[c]);
    }
}

// out[i,hc] = act(acc[i,hc]/(den[i,h]+EPS) + bias[hc]) ; act: 0=none, 1=elu
__global__ void k_attn_final(const float* __restrict__ acc,
                             const float* __restrict__ eden,
                             const float* __restrict__ bias,
                             float* __restrict__ out,
                             int n, int H, int C, int act) {
    long long idx = (long long)blockIdx.x * blockDim.x + threadIdx.x;
    long long total = (long long)n * H * C;
    if (idx >= total) return;
    int hc = (int)(idx % (H * C));
    int i  = (int)(idx / (H * C));
    int h  = hc / C;
    float v = acc[idx] / (eden[i * H + h] + EPS) + bias[hc];
    if (act == 1) v = v > 0.0f ? v : (__expf(v) - 1.0f);   // ELU
    out[idx] = v;
}

// gate = sigmoid([h2, hop] @ Wg + bg); h2 *= gate
__global__ void k_gate(float* __restrict__ h2, const float* __restrict__ hop,
                       const float* __restrict__ Wg, const float* __restrict__ bg,
                       int n) {
    int i = blockIdx.x * blockDim.x + threadIdx.x;
    if (i >= n) return;
    float* row = h2 + (long long)i * 128;
    float s = bg[0] + hop[i] * Wg[128];
    for (int j = 0; j < 128; ++j) s += row[j] * Wg[j];
    float g = 1.0f / (1.0f + __expf(-s));
    for (int j = 0; j < 128; ++j) row[j] *= g;
}

// final: bias + log_softmax over 40 classes -> d_out
__global__ void k_logsoftmax(const float* __restrict__ acc,
                             const float* __restrict__ eden,
                             const float* __restrict__ bias,
                             float* __restrict__ out, int n, int C) {
    int i = blockIdx.x * blockDim.x + threadIdx.x;
    if (i >= n) return;
    float v[40];
    float den = eden[i] + EPS;     // H == 1
    float m = -3.4e38f;
    for (int c = 0; c < C; ++c) {
        v[c] = acc[(long long)i * C + c] / den + bias[c];
        m = fmaxf(m, v[c]);
    }
    float s = 0.0f;
    for (int c = 0; c < C; ++c) s += __expf(v[c] - m);
    float ls = __logf(s);
    for (int c = 0; c < C; ++c) out[(long long)i * C + c] = v[c] - m - ls;
}

// ---------------------------------------------------------------------------
// Launch
// ---------------------------------------------------------------------------
extern "C" void kernel_launch(void* const* d_in, const int* in_sizes, int n_in,
                              void* d_out, int out_size, void* d_ws, size_t ws_size,
                              hipStream_t stream) {
    const float*     x    = (const float*)d_in[0];
    const long long* ei   = (const long long*)d_in[1];   // int64 [2,E]
    const int*       seed = (const int*)d_in[2];
    const float* W1  = (const float*)d_in[3];
    const float* a1s = (const float*)d_in[4];
    const float* a1d = (const float*)d_in[5];
    const float* b1  = (const float*)d_in[6];
    const float* W2  = (const float*)d_in[7];
    const float* a2s = (const float*)d_in[8];
    const float* a2d = (const float*)d_in[9];
    const float* b2  = (const float*)d_in[10];
    const float* W3  = (const float*)d_in[11];
    const float* a3s = (const float*)d_in[12];
    const float* a3d = (const float*)d_in[13];
    const float* b3  = (const float*)d_in[14];
    const float* Wg  = (const float*)d_in[15];
    const float* bg  = (const float*)d_in[16];
    float* out = (float*)d_out;

    const int N = in_sizes[0] / 128;          // 100000
    const int E = in_sizes[1] / 2;            // 1600000
    const long long* src = ei;
    const long long* dst = ei + E;
    const int Mt = (N + 15) / 16;             // 6250 row tiles

    // ---- workspace layout (bytes, 256-aligned chunks) --------------------
    char* ws = (char*)d_ws;
    size_t off = 0;
    auto take = [&](size_t bytes) -> void* {
        void* p = ws + off;
        off += (bytes + 255) & ~(size_t)255;
        return p;
    };
    int*   dist    = (int*)  take((size_t)N * 4);
    int*   reached = (int*)  take((size_t)N * 4);
    int*   nbr     = (int*)  take((size_t)N * 4);
    float* hop     = (float*)take((size_t)N * 4);
    float* aug     = (float*)take((size_t)N * 132 * 4);
    float* acc     = aug;                       // alias: aug dead after L1 pack
    float* ht      = (float*)take((size_t)Mt * 16 * 128 * 4);  // padded rows
    float* als     = (float*)take((size_t)N * 8 * 4);
    float* ald     = (float*)take((size_t)N * 8 * 4);
    float* emax    = (float*)take((size_t)N * 8 * 4);
    float* eden    = (float*)take((size_t)N * 8 * 4);
    float* h1      = (float*)take((size_t)N * 64 * 4);
    float* h2      = (float*)take((size_t)N * 128 * 4);
    _Float16* Ap   = (_Float16*)take((size_t)Mt * 5 * 512 * 2);  // max KT=5
    _Float16* Bp   = (_Float16*)take((size_t)8 * 5 * 512 * 2);   // max NT=8
    (void)ws_size; (void)n_in; (void)out_size;

    const int B = 256;
    const int nbN = (N + B - 1) / B;
    const int nbE = (E + B - 1) / B;
    auto blocks = [&](long long tot) { return (int)((tot + B - 1) / B); };

    // ---- BFS hierarchy ---------------------------------------------------
    k_bfs_init<<<nbN, B, 0, stream>>>(seed, dist, reached, nbr, N);
    for (int h = 1; h <= 3; ++h) {
        k_bfs_edge<<<nbE, B, 0, stream>>>(src, dst, reached, nbr, E);
        k_bfs_node<<<nbN, B, 0, stream>>>(dist, reached, nbr, h, N);
    }
    k_build_aug<<<blocks((long long)N * 132), B, 0, stream>>>(x, dist, aug, N);
    k_hop<<<nbN, B, 0, stream>>>(dist, hop, N);

    // ---- GAT layer 1: 132 -> 8 heads x 8 ch, ELU  (K=132 -> KT=5) --------
    {
        const int KT = 5, NC = 64, K = 132, NT = (NC + 15) / 16, ldc = NT * 16;
        k_pack_a<<<blocks((long long)Mt * KT * 512), B, 0, stream>>>(aug, Ap, N, K, KT);
        k_pack_b<<<blocks((long long)NT * KT * 512), B, 0, stream>>>(W1, Bp, K, NC, KT);
        dim3 g(Mt, NT);
        k_wmma_gemm<5><<<g, 32, 0, stream>>>(Ap, Bp, ht, ldc);
        k_attn_logits<<<nbN, B, 0, stream>>>(ht, a1s, a1d, als, ald, N, 8, 8, ldc);
        k_attn_init<<<nbN, B, 0, stream>>>(als, ald, emax, N, 8);
        k_attn_edge_max<<<nbE, B, 0, stream>>>(src, dst, als, ald, emax, E, 8);
        k_attn_self<<<nbN, B, 0, stream>>>(ht, als, ald, emax, eden, acc, N, 8, 8, ldc);
        k_attn_edge_sum<<<nbE, B, 0, stream>>>(src, dst, ht, als, ald, emax, eden, acc, E, 8, 8, ldc);
        k_attn_final<<<blocks((long long)N * 64), B, 0, stream>>>(acc, eden, b1, h1, N, 8, 8, /*elu*/1);
    }

    // ---- GAT layer 2: 64 -> 8 heads x 16 ch  (K=64 -> KT=2) --------------
    {
        const int KT = 2, NC = 128, K = 64, NT = (NC + 15) / 16, ldc = NT * 16;
        k_pack_a<<<blocks((long long)Mt * KT * 512), B, 0, stream>>>(h1, Ap, N, K, KT);
        k_pack_b<<<blocks((long long)NT * KT * 512), B, 0, stream>>>(W2, Bp, K, NC, KT);
        dim3 g(Mt, NT);
        k_wmma_gemm<2><<<g, 32, 0, stream>>>(Ap, Bp, ht, ldc);
        k_attn_logits<<<nbN, B, 0, stream>>>(ht, a2s, a2d, als, ald, N, 8, 16, ldc);
        k_attn_init<<<nbN, B, 0, stream>>>(als, ald, emax, N, 8);
        k_attn_edge_max<<<nbE, B, 0, stream>>>(src, dst, als, ald, emax, E, 8);
        k_attn_self<<<nbN, B, 0, stream>>>(ht, als, ald, emax, eden, acc, N, 8, 16, ldc);
        k_attn_edge_sum<<<nbE, B, 0, stream>>>(src, dst, ht, als, ald, emax, eden, acc, E, 8, 16, ldc);
        k_attn_final<<<blocks((long long)N * 128), B, 0, stream>>>(acc, eden, b2, h2, N, 8, 16, /*none*/0);
    }

    // ---- gate ------------------------------------------------------------
    k_gate<<<nbN, B, 0, stream>>>(h2, hop, Wg, bg, N);

    // ---- GAT layer 3: 128 -> 1 head x 40 ch, log_softmax (K=128 -> KT=4) -
    {
        const int KT = 4, NC = 40, K = 128, NT = (NC + 15) / 16, ldc = NT * 16;
        k_pack_a<<<blocks((long long)Mt * KT * 512), B, 0, stream>>>(h2, Ap, N, K, KT);
        k_pack_b<<<blocks((long long)NT * KT * 512), B, 0, stream>>>(W3, Bp, K, NC, KT);
        dim3 g(Mt, NT);
        k_wmma_gemm<4><<<g, 32, 0, stream>>>(Ap, Bp, ht, ldc);
        k_attn_logits<<<nbN, B, 0, stream>>>(ht, a3s, a3d, als, ald, N, 1, 40, ldc);
        k_attn_init<<<nbN, B, 0, stream>>>(als, ald, emax, N, 1);
        k_attn_edge_max<<<nbE, B, 0, stream>>>(src, dst, als, ald, emax, E, 1);
        k_attn_self<<<nbN, B, 0, stream>>>(ht, als, ald, emax, eden, acc, N, 1, 40, ldc);
        k_attn_edge_sum<<<nbE, B, 0, stream>>>(src, dst, ht, als, ald, emax, eden, acc, E, 1, 40, ldc);
        k_logsoftmax<<<nbN, B, 0, stream>>>(acc, eden, b3, out, N, 40);
    }
}
